// POSTagEncoder_13142599926348
// MI455X (gfx1250) — compile-verified, matrix operand order checked by reading, not run on hardware
//
#include <hip/hip_runtime.h>
#include <hip/hip_bf16.h>
#include <math.h>

#define S_LEN 8192
#define DIM   128
#define NEXP  48
#define G4    512   // 4*DIM
#define WSTR  132   // padded LSTM weight row stride (floats): 528B, 16B-aligned

typedef __bf16 bf16_t;
typedef __attribute__((ext_vector_type(16))) __bf16    v16bf;
typedef __attribute__((ext_vector_type(8)))  float     v8f;
typedef __attribute__((ext_vector_type(4)))  float     f32x4;
typedef __attribute__((ext_vector_type(8)))  unsigned  u32x8;

// ---------- cheap bf16 hi/lo split, packed two-at-a-time ----------
// hi = truncate-to-bf16; lo = exact residual, truncate-packed.
// hi*hi + hi*lo + lo*hi carries ~16+ mantissa bits (tf32x3-class accuracy).
// Pack = single V_PERM_B32: dst = {b[3],b[2],a[3],a[2]}  (sel 0-3 -> src1).
__device__ inline unsigned pkhi(float a, float b) {
  return __builtin_amdgcn_perm(__float_as_uint(b), __float_as_uint(a), 0x07060302u);
}
__device__ inline float resid(float a) {
  return a - __uint_as_float(__float_as_uint(a) & 0xffff0000u);
}

// Build one 16x32 bf16 fragment pair (hi, lo) for this lane from a row-major
// f32 row. ISA 7.12.2 16-bit A/B layout: VGPR j holds K pair {k,k+1} with
// K = (j&3)*2 + 8*hsel + (j>>2)*16  => per lane two contiguous 8-float runs.
__device__ inline void build_frag(const float* __restrict__ row, int kb, int hsel,
                                  v16bf& hi, v16bf& lo) {
  const float* p = row + kb + 8 * hsel;
  f32x4 q0 = *(const f32x4*)(p);
  f32x4 q1 = *(const f32x4*)(p + 4);
  f32x4 q2 = *(const f32x4*)(p + 16);
  f32x4 q3 = *(const f32x4*)(p + 20);
  u32x8 h, l;
  h[0] = pkhi(q0.x, q0.y); h[1] = pkhi(q0.z, q0.w);
  h[2] = pkhi(q1.x, q1.y); h[3] = pkhi(q1.z, q1.w);
  h[4] = pkhi(q2.x, q2.y); h[5] = pkhi(q2.z, q2.w);
  h[6] = pkhi(q3.x, q3.y); h[7] = pkhi(q3.z, q3.w);
  l[0] = pkhi(resid(q0.x), resid(q0.y)); l[1] = pkhi(resid(q0.z), resid(q0.w));
  l[2] = pkhi(resid(q1.x), resid(q1.y)); l[3] = pkhi(resid(q1.z), resid(q1.w));
  l[4] = pkhi(resid(q2.x), resid(q2.y)); l[5] = pkhi(resid(q2.z), resid(q2.w));
  l[6] = pkhi(resid(q3.x), resid(q3.y)); l[7] = pkhi(resid(q3.z), resid(q3.w));
  hi = __builtin_bit_cast(v16bf, h);
  lo = __builtin_bit_cast(v16bf, l);
}

// ---------- phase 0: bucket tokens by expert ----------
__global__ void bucket_kernel(const int* __restrict__ pos_ids,
                              int* __restrict__ counts,
                              int* __restrict__ lists) {
  int s = blockIdx.x * blockDim.x + threadIdx.x;
  if (s < S_LEN) {
    int e = pos_ids[s];
    int p = atomicAdd(&counts[e], 1);
    lists[(size_t)e * S_LEN + p] = s;
  }
}

// ---------- phase 1: per-expert GEMM  z = X_e @ W_e^T + b_e  (bf16x3 WMMA) ----------
__global__ __launch_bounds__(256)
void expert_gemm_kernel(const float* __restrict__ embs,
                        const float* __restrict__ W_exp,
                        const float* __restrict__ b_exp,
                        const int*   __restrict__ counts,
                        const int*   __restrict__ lists,
                        float*       __restrict__ z) {
  const int e   = blockIdx.x;
  const int cnt = counts[e];
  if (cnt == 0) return;
  const int wave = threadIdx.x >> 5;   // 0..7 -> N-tile of 16 columns
  const int lane = threadIdx.x & 31;
  const int hsel = lane >> 4;
  const int m    = lane & 15;
  const int n0   = wave * 16;
  const float* W   = W_exp + (size_t)e * DIM * DIM;
  const int*   lst = lists + (size_t)e * S_LEN;

  // B[k][n] = W[n][k]; per-lane source row is W[n0+m][*]. Built once, kept in regs.
  v16bf bhi[4], blo[4];
  #pragma unroll
  for (int kc = 0; kc < 4; ++kc)
    build_frag(W + (size_t)(n0 + m) * DIM, kc * 32, hsel, bhi[kc], blo[kc]);
  const float bias = b_exp[e * DIM + n0 + m];

  for (int base = 0; base < cnt; base += 16) {
    int myrow = base + m; if (myrow >= cnt) myrow = cnt - 1;
    const int tok = lst[myrow];
    const float* xrow = embs + (size_t)tok * DIM;
    v8f acc;
    #pragma unroll
    for (int r = 0; r < 8; ++r) acc[r] = bias;   // bias depends only on column N
    #pragma unroll
    for (int kc = 0; kc < 4; ++kc) {
      v16bf ahi, alo;
      build_frag(xrow, kc * 32, hsel, ahi, alo);
      acc = __builtin_amdgcn_wmma_f32_16x16x32_bf16(false, ahi, false, bhi[kc], (short)0, acc, false, false);
      acc = __builtin_amdgcn_wmma_f32_16x16x32_bf16(false, ahi, false, blo[kc], (short)0, acc, false, false);
      acc = __builtin_amdgcn_wmma_f32_16x16x32_bf16(false, alo, false, bhi[kc], (short)0, acc, false, false);
    }
    #pragma unroll
    for (int r = 0; r < 8; ++r) {      // D: M = r + 8*hsel, N = m
      int row = base + r + 8 * hsel;
      if (row < cnt) {
        int t2 = lst[row];
        z[(size_t)t2 * DIM + n0 + m] = acc[r];
      }
    }
  }
}

// ---------- phase 2: x_gates = z @ W_ih^T + (b_ih + b_hh)  (bf16x3 WMMA) ----------
__global__ __launch_bounds__(256)
void xgates_gemm_kernel(const float* __restrict__ z,
                        const float* __restrict__ W_ih,
                        const float* __restrict__ b_ih,
                        const float* __restrict__ b_hh,
                        float*       __restrict__ xg) {
  const int wave = threadIdx.x >> 5;
  const int lane = threadIdx.x & 31;
  const int hsel = lane >> 4, m = lane & 15;
  const int ct = blockIdx.x & 31;      // column tile 0..31 (512 gate outputs)
  const int rp = blockIdx.x >> 5;      // row partition 0..7
  const int n  = ct * 16 + m;

  v16bf bhi[4], blo[4];                // B[k][n] = W_ih[n][k]; built once, reused 8x
  #pragma unroll
  for (int kc = 0; kc < 4; ++kc)
    build_frag(W_ih + (size_t)n * DIM, kc * 32, hsel, bhi[kc], blo[kc]);
  const float bias = b_ih[n] + b_hh[n];

  for (int rt = rp * 64 + wave; rt < rp * 64 + 64; rt += 8) {
    const int s0 = rt * 16;
    const float* arow = z + (size_t)(s0 + m) * DIM;
    v8f acc;
    #pragma unroll
    for (int r = 0; r < 8; ++r) acc[r] = bias;
    #pragma unroll
    for (int kc = 0; kc < 4; ++kc) {
      v16bf ahi, alo;
      build_frag(arow, kc * 32, hsel, ahi, alo);
      acc = __builtin_amdgcn_wmma_f32_16x16x32_bf16(false, ahi, false, bhi[kc], (short)0, acc, false, false);
      acc = __builtin_amdgcn_wmma_f32_16x16x32_bf16(false, ahi, false, blo[kc], (short)0, acc, false, false);
      acc = __builtin_amdgcn_wmma_f32_16x16x32_bf16(false, alo, false, bhi[kc], (short)0, acc, false, false);
    }
    #pragma unroll
    for (int r = 0; r < 8; ++r)
      xg[(size_t)(s0 + r + 8 * hsel) * G4 + n] = acc[r];
  }
}

// ---------- phase 3: sequential LSTM, one persistent WGP, W_hh in 320KB LDS ----------
__global__ __launch_bounds__(512)
void lstm_kernel(const float* __restrict__ xg,
                 const float* __restrict__ W_hh,
                 float*       __restrict__ out) {
  extern __shared__ float smem[];
  float* Wp    = smem;                  // [512][WSTR] padded row-major -> ds_load_b128
  float* hbuf  = smem + G4 * WSTR;      // [128]
  float* gates = hbuf + DIM;            // [512]
  const int t = threadIdx.x;            // one lane per gate output

  #pragma unroll
  for (int k = 0; k < DIM; k += 4)      // copy own row, vectorized both sides
    *(f32x4*)&Wp[t * WSTR + k] = *(const f32x4*)&W_hh[(size_t)t * DIM + k];
  if (t < DIM) hbuf[t] = 0.0f;
  __syncthreads();

  float c = 0.0f, hmax = -1e30f;        // live only for t < 128
  const float* xp = xg + t;
  for (int s = 0; s < S_LEN; ++s) {
    float acc = xp[(size_t)s * G4];
    if (s + 1 < S_LEN) __builtin_prefetch(&xp[(size_t)(s + 1) * G4], 0, 0);
    #pragma unroll
    for (int k = 0; k < DIM; k += 4) {  // b128 W load + b128 h broadcast + 4 fma
      f32x4 w  = *(const f32x4*)&Wp[t * WSTR + k];
      f32x4 hv = *(const f32x4*)&hbuf[k];
      acc = fmaf(hv.x, w.x, acc);
      acc = fmaf(hv.y, w.y, acc);
      acc = fmaf(hv.z, w.z, acc);
      acc = fmaf(hv.w, w.w, acc);
    }
    gates[t] = acc;
    __syncthreads();
    if (t < DIM) {
      float gi = gates[t], gf = gates[DIM + t], gg = gates[2 * DIM + t], go = gates[3 * DIM + t];
      gi = 1.0f / (1.0f + expf(-gi));
      gf = 1.0f / (1.0f + expf(-gf));
      gg = tanhf(gg);
      go = 1.0f / (1.0f + expf(-go));
      c = gf * c + gi * gg;
      float hn = go * tanhf(c);
      hbuf[t] = hn;
      hmax = fmaxf(hmax, hn);
    }
    __syncthreads();
  }
  if (t < DIM) out[t] = hmax;
}

// ---------- host ----------
extern "C" void kernel_launch(void* const* d_in, const int* in_sizes, int n_in,
                              void* d_out, int out_size, void* d_ws, size_t ws_size,
                              hipStream_t stream) {
  (void)in_sizes; (void)n_in; (void)out_size;
  const float* embs  = (const float*)d_in[0];
  const int*   pos   = (const int*)  d_in[1];
  const float* W_exp = (const float*)d_in[2];
  const float* b_exp = (const float*)d_in[3];
  const float* W_ih  = (const float*)d_in[4];
  const float* W_hh  = (const float*)d_in[5];
  const float* b_ih  = (const float*)d_in[6];
  const float* b_hh  = (const float*)d_in[7];
  float* out = (float*)d_out;

  const size_t zBytes   = (size_t)S_LEN * DIM * sizeof(float);   // 4 MB
  const size_t xgBytes  = (size_t)S_LEN * G4  * sizeof(float);   // 16 MB
  const size_t cntBytes = 64 * sizeof(int);
  const size_t lstBytes = (size_t)NEXP * S_LEN * sizeof(int);    // 1.5 MB
  if (ws_size < zBytes + xgBytes + cntBytes + lstBytes) return;

  char* ws = (char*)d_ws;
  float* z      = (float*)ws;
  float* xg     = (float*)(ws + zBytes);
  int*   counts = (int*)  (ws + zBytes + xgBytes);
  int*   lists  = counts + 64;

  hipMemsetAsync(counts, 0, cntBytes, stream);
  bucket_kernel<<<(S_LEN + 255) / 256, 256, 0, stream>>>(pos, counts, lists);
  expert_gemm_kernel<<<NEXP, 256, 0, stream>>>(embs, W_exp, b_exp, counts, lists, z);
  xgates_gemm_kernel<<<256, 256, 0, stream>>>(z, W_ih, b_ih, b_hh, xg);

  const size_t ldsBytes = (size_t)(G4 * WSTR + DIM + G4) * sizeof(float); // ~267 KB < 320 KB
  hipFuncSetAttribute(reinterpret_cast<const void*>(lstm_kernel),
                      hipFuncAttributeMaxDynamicSharedMemorySize, (int)ldsBytes);
  lstm_kernel<<<1, 512, ldsBytes, stream>>>(xg, W_hh, out);
}
